// SPP_CNN_13305808683529
// MI455X (gfx1250) — matrix operand order repaired and checked by reading, not compiled, not run on hardware
//
#include <hip/hip_runtime.h>

// ---------------------------------------------------------------------------
// SPP-CNN fused pipeline for MI455X (gfx1250, wave32, WMMA)
//   conv1d(K=21) as 21 shifted GEMMs on v_wmma_f32_16x16x32_f16.
//   Per kc-chunk the schedule is pinned with sched_group_barrier to a
//   2-deep software pipeline: [2 VMEM rd (next A)] [4 DS rd] then
//   (1 WMMA : 2 DS rd) x6 then 2 WMMA — B fragments stay in flight ahead
//   of the consuming WMMA instead of load->wait(0)->wmma lockstep.
//   Fused BN + LeakyReLU + ragged SPP max (order-encoded atomicMax) + FC.
// ---------------------------------------------------------------------------

typedef _Float16 v8h  __attribute__((ext_vector_type(8)));
typedef _Float16 v16h __attribute__((ext_vector_type(16)));
typedef float    v8f  __attribute__((ext_vector_type(8)));

#define BSZ     32
#define CCH     128
#define T_IN    8192
#define KW      21
#define T_OUT   (T_IN - KW + 1)      // 8172
#define TTILE   128
#define NTT     64                   // ceil(T_OUT / TTILE)
#define LDS_T   (TTILE + KW - 1)     // 148 rows (t-halo)
#define LDS_CS  136                  // halfs per row: 128 + 8 pad (272B)
#define NFEAT   768                  // 6*C pooled features
#define WBLK    512                  // halfs per (k, co-tile, ci-chunk) A block
#define NWBLK   (KW * 8 * 4)         // 672 blocks, 344064 halfs total

#if defined(__has_builtin) && __has_builtin(__builtin_amdgcn_sched_group_barrier)
#define SGB(mask, size, id) __builtin_amdgcn_sched_group_barrier((mask), (size), (id))
#define SCHED_FENCE()
#else
#define SGB(mask, size, id)
#define SCHED_FENCE() asm volatile("" ::: "memory")
#endif

// Monotone float <-> uint encoding so atomicMax(u32) == float max (handles sign).
__device__ __forceinline__ unsigned enc_f(float f) {
    unsigned u = __float_as_uint(f);
    return (f >= 0.0f) ? (u | 0x80000000u) : ~u;
}
__device__ __forceinline__ float dec_f(unsigned u) {
    return (u & 0x80000000u) ? __uint_as_float(u ^ 0x80000000u)
                             : __uint_as_float(~u);
}

__device__ __forceinline__ v16h cat16(v8h lo, v8h hi) {
    return __builtin_shufflevector(lo, hi,
        0,1,2,3,4,5,6,7,8,9,10,11,12,13,14,15);
}

// -------- kernel 0: clear pooled-feature buffer (identity of encoded max) ---
__global__ __launch_bounds__(256) void spp_init_kernel(unsigned* __restrict__ fbuf) {
    int i = blockIdx.x * 256 + threadIdx.x;
    if (i < BSZ * NFEAT) fbuf[i] = 0u;
}

// -------- kernel 1: swizzle conv weights into WMMA-A lane layout (f16) ------
__global__ __launch_bounds__(256) void wprep_kernel(const float* __restrict__ cw,
                                                    _Float16* __restrict__ wbuf) {
    int idx = blockIdx.x * 256 + threadIdx.x;          // 344064 total
    if (idx >= NWBLK * WBLK) return;
    int blk = idx >> 9;
    int wi  = idx & 511;
    int l   = wi >> 4;
    int h   = wi & 15;
    int k   = blk >> 5;
    int rem = blk & 31;
    int mt  = rem >> 2;
    int kc  = rem & 3;
    int co  = mt * 16 + (l & 15);
    int g   = l >> 4;
    int ci  = kc * 32 + g * 16 + h;                    // A mapping: contiguous k = 16*g + h
    wbuf[idx] = (_Float16)cw[(co * CCH + ci) * KW + k];
}

// -------- kernel 2: fused conv(WMMA) + BN + LeakyReLU + SPP atomic max ------
__global__ __launch_bounds__(256) void conv_spp_kernel(
    const float* __restrict__ x, const int* __restrict__ olen,
    const _Float16* __restrict__ wbuf,
    const float* __restrict__ gamma, const float* __restrict__ beta,
    const float* __restrict__ mean,  const float* __restrict__ var,
    unsigned* __restrict__ fbuf)
{
    __shared__ _Float16 xs[LDS_T * LDS_CS];            // 40256 B
    const int b   = blockIdx.y;
    const int t0  = blockIdx.x * TTILE;
    const int tid = threadIdx.x;

    // Stage x[b, :, t0 .. t0+147] into LDS as f16, t-major rows of ci.
    const float* xb = x + (size_t)b * CCH * T_IN;
    for (int idx = tid; idx < CCH * LDS_T; idx += 256) {
        int ci = idx / LDS_T;
        int tt = idx - ci * LDS_T;
        int t  = t0 + tt;
        float v = (t < T_IN) ? xb[(size_t)ci * T_IN + t] : 0.0f;
        xs[tt * LDS_CS + ci] = (_Float16)v;
    }
    __syncthreads();

    const int w    = tid >> 5;     // wave id == co tile (0..7)
    const int lane = tid & 31;
    const int g    = lane >> 4;
    const int n    = lane & 15;

    auto load_a = [&](int k, int kc) -> v16h {
        const _Float16* ap =
            wbuf + ((size_t)(k * 8 + w) * 4 + kc) * WBLK + lane * 16;
        return cat16(*(const v8h*)ap, *(const v8h*)(ap + 8));
    };
    auto load_b = [&](int k, int kc, int nt) -> v16h {
        const _Float16* bp =
            xs + (nt * 16 + n + k) * LDS_CS + (kc * 32 + g * 16);
        return cat16(*(const v8h*)bp, *(const v8h*)(bp + 8));
    };

    v8f acc[8] = {};
    v16h a = load_a(0, 0);         // A double-buffered one kc-chunk ahead

    #pragma unroll 1
    for (int k = 0; k < KW; ++k) {
        if (k + 1 < KW)            // pull next tap's A slice toward L2/L0
            __builtin_prefetch(
                wbuf + (size_t)((k + 1) * 8 + w) * 4 * WBLK, 0, 1);
        #pragma unroll
        for (int kc = 0; kc < 4; ++kc) {
            // Next A fragment in flight while this chunk's WMMAs run.
            v16h an = (kc < 3) ? load_a(k, kc + 1)
                               : load_a((k + 1 < KW) ? k + 1 : k, 0);
            v16h bf0 = load_b(k, kc, 0);
            v16h bf1 = load_b(k, kc, 1);
            v16h bf2 = load_b(k, kc, 2);
            v16h bf3 = load_b(k, kc, 3);
            v16h bf4 = load_b(k, kc, 4);
            v16h bf5 = load_b(k, kc, 5);
            v16h bf6 = load_b(k, kc, 6);
            v16h bf7 = load_b(k, kc, 7);
            SCHED_FENCE();
            acc[0] = __builtin_amdgcn_wmma_f32_16x16x32_f16(
                false, a, false, bf0, (short)0, acc[0], false, false);
            acc[1] = __builtin_amdgcn_wmma_f32_16x16x32_f16(
                false, a, false, bf1, (short)0, acc[1], false, false);
            acc[2] = __builtin_amdgcn_wmma_f32_16x16x32_f16(
                false, a, false, bf2, (short)0, acc[2], false, false);
            acc[3] = __builtin_amdgcn_wmma_f32_16x16x32_f16(
                false, a, false, bf3, (short)0, acc[3], false, false);
            acc[4] = __builtin_amdgcn_wmma_f32_16x16x32_f16(
                false, a, false, bf4, (short)0, acc[4], false, false);
            acc[5] = __builtin_amdgcn_wmma_f32_16x16x32_f16(
                false, a, false, bf5, (short)0, acc[5], false, false);
            acc[6] = __builtin_amdgcn_wmma_f32_16x16x32_f16(
                false, a, false, bf6, (short)0, acc[6], false, false);
            acc[7] = __builtin_amdgcn_wmma_f32_16x16x32_f16(
                false, a, false, bf7, (short)0, acc[7], false, false);
            a = an;
            // Pin the per-chunk schedule: A loads first, B frags 0-1, then
            // WMMA_i overlapped with the DS reads of fragment i+2.
            SGB(0x020, 2, 0);               // 2 VMEM reads  (next A)
            SGB(0x100, 4, 0);               // 4 DS reads    (B frag 0,1)
            #pragma unroll
            for (int i = 0; i < 6; ++i) {
                SGB(0x008, 1, 0);           // 1 WMMA        (consumes frag i)
                SGB(0x100, 2, 0);           // 2 DS reads    (B frag i+2)
            }
            SGB(0x008, 2, 0);               // final 2 WMMAs
        }
    }

    // Epilogue: BN + LeakyReLU, then SPP max straight from accumulators.
    const int len = olen[b];
    unsigned* fb = fbuf + b * NFEAT;
    #pragma unroll 1
    for (int r = 0; r < 8; ++r) {
        int   m  = r + g * 8;                 // C/D layout: VGPR r, halfgroup g
        int   co = w * 16 + m;
        float sc = gamma[co] * rsqrtf(var[co] + 1e-5f);
        float bi = beta[co] - mean[co] * sc;
        float val[8];
        #pragma unroll
        for (int nt = 0; nt < 8; ++nt) {
            float y = sc * acc[nt][r] + bi;
            val[nt] = (y >= 0.0f) ? y : 0.01f * y;
        }
        #pragma unroll 1
        for (int sh = 0; sh < 2; ++sh) {
            int L    = len - (sh ? 40 : 20);  // shift = (K-1), 2*(K-1)
            int s    = L >> 1;                // lvl1 stride  = floor(L/2)
            int kk   = (L + 1) >> 1;          // lvl1 kernel  = ceil(L/2)
            int base = sh * 384;              // 3*C per spp branch
            #pragma unroll 1
            for (int wd = 0; wd < 3; ++wd) {  // {lvl0, lvl1 j=0, lvl1 j=1}
                int lo = (wd == 2) ? s : 0;
                int hi = (wd == 1) ? kk : L;
                float mx = -3.4e38f;
                #pragma unroll
                for (int nt = 0; nt < 8; ++nt) {
                    int t = t0 + nt * 16 + n;
                    bool ok = (t < T_OUT) & (t >= lo) & (t < hi);
                    mx = ok ? fmaxf(mx, val[nt]) : mx;
                }
                #pragma unroll
                for (int d = 8; d >= 1; d >>= 1)  // reduce over 16-lane group
                    mx = fmaxf(mx, __shfl_xor(mx, d, 32));
                if (n == 0 && mx > -3.0e38f) {
                    int idx = (wd == 0) ? (base + co)
                                        : (base + 128 + co * 2 + (wd - 1));
                    atomicMax(fb + idx, enc_f(mx));
                }
            }
        }
    }
}

// -------- kernel 3: decode pooled features, FC to [32, 2] -------------------
__global__ __launch_bounds__(64) void fc_kernel(const unsigned* __restrict__ fbuf,
                                                const float* __restrict__ fcw,
                                                const float* __restrict__ fcb,
                                                float* __restrict__ out) {
    int b    = blockIdx.x;
    int task = threadIdx.x >> 5;
    int lane = threadIdx.x & 31;
    const unsigned* fb  = fbuf + b * NFEAT;
    const float*   wrow = fcw + task * NFEAT;
    float s = 0.0f;
    for (int i = lane; i < NFEAT; i += 32)
        s += dec_f(fb[i]) * wrow[i];
    #pragma unroll
    for (int d = 16; d >= 1; d >>= 1) s += __shfl_xor(s, d, 32);
    if (lane == 0) out[b * 2 + task] = s + fcb[task];
}

extern "C" void kernel_launch(void* const* d_in, const int* in_sizes, int n_in,
                              void* d_out, int out_size, void* d_ws, size_t ws_size,
                              hipStream_t stream) {
    const float* x    = (const float*)d_in[0];
    const int*   olen = (const int*)  d_in[1];
    const float* cw   = (const float*)d_in[2];
    const float* gam  = (const float*)d_in[3];
    const float* bet  = (const float*)d_in[4];
    const float* mn   = (const float*)d_in[5];
    const float* vr   = (const float*)d_in[6];
    const float* fcw  = (const float*)d_in[7];
    const float* fcb  = (const float*)d_in[8];

    unsigned* fbuf = (unsigned*)d_ws;                          // 98304 B
    _Float16* wbuf = (_Float16*)((char*)d_ws + 98304);         // 688128 B

    spp_init_kernel<<<(BSZ * NFEAT + 255) / 256, 256, 0, stream>>>(fbuf);
    wprep_kernel<<<(NWBLK * WBLK + 255) / 256, 256, 0, stream>>>(cw, wbuf);
    dim3 grid(NTT, BSZ);
    conv_spp_kernel<<<grid, 256, 0, stream>>>(x, olen, wbuf,
                                              gam, bet, mn, vr, fbuf);
    fc_kernel<<<BSZ, 64, 0, stream>>>(fbuf, fcw, fcb, (float*)d_out);
}
